// DotProductAttention_79774722555941
// MI455X (gfx1250) — matrix operand order repaired and checked by reading, not compiled, not run on hardware
//
#include <hip/hip_runtime.h>

typedef __attribute__((ext_vector_type(16))) _Float16 v16h;
typedef __attribute__((ext_vector_type(8)))  _Float16 v8h;
typedef __attribute__((ext_vector_type(8)))  float    v8f;
typedef __attribute__((ext_vector_type(4)))  float    v4f;

#define DK    64
#define BQ    128   // q rows per workgroup (8 waves x 16)
#define BKV   32    // kv rows per tile
#define KPAD  72    // Ks row stride in halves (144B, 16B aligned)
#define VPAD  40    // Vst row stride in halves (80B, 16B aligned)

#define NBATCH 32
#define SEQLEN 2048
#define NTILES (SEQLEN / BKV)

__global__ __launch_bounds__(256)
void flash_attn_f16wmma(const float* __restrict__ Qg,
                        const float* __restrict__ Kg,
                        const float* __restrict__ Vg,
                        float* __restrict__ Og) {
  // Double-buffered tiles: compute from buf while filling buf^1.
  __shared__ alignas(16) _Float16 Ks [2][BKV][KPAD];  // K tile, row-major [kv][d]
  __shared__ alignas(16) _Float16 Vst[2][DK ][VPAD];  // V tile, transposed [d][kv]

  const int tid  = threadIdx.x;
  const int lane = tid & 31;
  const int hf   = lane >> 4;    // which 16-lane half
  const int ln16 = lane & 15;

  const int b     = blockIdx.y;
  const int qbase = blockIdx.x * BQ + (tid >> 5) * 16; // this wave's 16 q rows
  const int qrow  = qbase + ln16;

  const float scale = 0.125f; // 1/sqrt(64), folded into Q

  // ---- Load Q^T as two B-fragments (kk = i + 16*hf + 32*s): 16 contiguous f32.
  // Q is touched exactly once per element -> nontemporal loads.
  v16h QB[2];
  {
    const float* qp = Qg + ((size_t)b * SEQLEN + qrow) * DK;
#pragma unroll
    for (int s = 0; s < 2; ++s) {
      const v4f* q4 = (const v4f*)(qp + 16 * hf + 32 * s);
#pragma unroll
      for (int j = 0; j < 4; ++j) {
        v4f f = __builtin_nontemporal_load(q4 + j);
        QB[s][4 * j + 0] = (_Float16)(f.x * scale);
        QB[s][4 * j + 1] = (_Float16)(f.y * scale);
        QB[s][4 * j + 2] = (_Float16)(f.z * scale);
        QB[s][4 * j + 3] = (_Float16)(f.w * scale);
      }
    }
  }

  v8f Oacc[4] = {v8f{}, v8f{}, v8f{}, v8f{}}; // O tile 16(q) x 64(d), f32 acc
  float mrun = -3.0e38f;  // running row max (per lane: q = ln16)
  float lrun = 0.0f;      // running row sum

  const size_t kvstride = (size_t)b * SEQLEN * DK;

  // Staging indices for this thread: 2 float4s per matrix per tile.
  int sr[2], sc4[2];
#pragma unroll
  for (int j = 0; j < 2; ++j) {
    int iv = j * 256 + tid;   // 512 float4s = 32x64 f32
    sr[j]  = iv >> 4;         // kv row 0..31
    sc4[j] = iv & 15;         // float4 col 0..15
  }

  v4f pk[2], pv[2]; // prefetch registers (global -> reg -> LDS pipeline)

  // ---- Prologue: fetch + stage tile 0
#pragma unroll
  for (int j = 0; j < 2; ++j) {
    size_t goff = kvstride + (size_t)(sr[j]) * DK + sc4[j] * 4;
    pk[j] = *(const v4f*)(Kg + goff);
    pv[j] = *(const v4f*)(Vg + goff);
  }
#pragma unroll
  for (int j = 0; j < 2; ++j) {
    int r = sr[j], c = sc4[j] * 4;
    Ks[0][r][c + 0] = (_Float16)pk[j].x;
    Ks[0][r][c + 1] = (_Float16)pk[j].y;
    Ks[0][r][c + 2] = (_Float16)pk[j].z;
    Ks[0][r][c + 3] = (_Float16)pk[j].w;
    Vst[0][c + 0][r] = (_Float16)pv[j].x;
    Vst[0][c + 1][r] = (_Float16)pv[j].y;
    Vst[0][c + 2][r] = (_Float16)pv[j].z;
    Vst[0][c + 3][r] = (_Float16)pv[j].w;
  }
  __syncthreads();

  int buf = 0;
  for (int kt = 0; kt < NTILES; ++kt) {
    // ---- Prefetch next tile from global while this tile computes (hide latency)
    const bool more = (kt + 1) < NTILES;
    if (more) {
      const int kvnext = (kt + 1) * BKV;
#pragma unroll
      for (int j = 0; j < 2; ++j) {
        size_t goff = kvstride + (size_t)(kvnext + sr[j]) * DK + sc4[j] * 4;
        pk[j] = *(const v4f*)(Kg + goff);
        pv[j] = *(const v4f*)(Vg + goff);
      }
    }

    // ---- S^T = K_tile (A, 2 frags of 16 kv rows) x Q^T (B). D: M=kv, N=q.
    v8f st[2];
#pragma unroll
    for (int f = 0; f < 2; ++f) {
      v8f acc = v8f{};
#pragma unroll
      for (int ks = 0; ks < 2; ++ks) {
        // A-layout: K = i + 16*hf + 32*ks ; M = f*16 + ln16
        const _Float16* row = &Ks[buf][f * 16 + ln16][0];
        v8h c0 = *(const v8h*)(row + 16 * hf + 32 * ks);
        v8h c1 = *(const v8h*)(row + 16 * hf + 8 + 32 * ks);
        v16h KA = __builtin_shufflevector(c0, c1, 0, 1, 2, 3, 4, 5, 6, 7,
                                          8, 9, 10, 11, 12, 13, 14, 15);
        acc = __builtin_amdgcn_wmma_f32_16x16x32_f16(
            false, KA, false, QB[ks], (short)0, acc, false, false);
      }
      st[f] = acc;
    }

    // ---- Online softmax (lane owns q = ln16; halves merged via shfl_xor 16)
    float mx = -3.0e38f;
#pragma unroll
    for (int f = 0; f < 2; ++f)
#pragma unroll
      for (int j = 0; j < 8; ++j) mx = fmaxf(mx, st[f][j]);
    mx = fmaxf(mx, __shfl_xor(mx, 16, 32));

    float mnew  = fmaxf(mrun, mx);
    float alpha = __expf(mrun - mnew);
    float rs = 0.0f;
#pragma unroll
    for (int f = 0; f < 2; ++f)
#pragma unroll
      for (int j = 0; j < 8; ++j) {
        float p = __expf(st[f][j] - mnew);
        st[f][j] = p;
        rs += p;
      }
    rs += __shfl_xor(rs, 16, 32);
    lrun = lrun * alpha + rs;
    mrun = mnew;

    // ---- Rescale O (O-lane holds q = v + 8*hf; alpha for q=r lives in lane r)
#pragma unroll
    for (int v = 0; v < 8; ++v) {
      float av = __shfl(alpha, (hf << 3) + v, 32);
#pragma unroll
      for (int dt = 0; dt < 4; ++dt) Oacc[dt][v] *= av;
    }

    // ---- P (S^T C-layout) -> A-layout: pure per-lane register remap
    v16h PA;
#pragma unroll
    for (int i = 0; i < 16; ++i) PA[i] = (_Float16)st[i / 8][i % 8];

    // ---- O += P x V  (B from transposed V in LDS: kk = (i%8) + 16*(i/8) + 8*hf)
#pragma unroll
    for (int dt = 0; dt < 4; ++dt) {
      const _Float16* vr = &Vst[buf][dt * 16 + ln16][0];
      v8h b0 = *(const v8h*)(vr + 8 * hf);
      v8h b1 = *(const v8h*)(vr + 16 + 8 * hf);
      v16h VB = __builtin_shufflevector(b0, b1, 0, 1, 2, 3, 4, 5, 6, 7,
                                        8, 9, 10, 11, 12, 13, 14, 15);
      Oacc[dt] = __builtin_amdgcn_wmma_f32_16x16x32_f16(
          false, PA, false, VB, (short)0, Oacc[dt], false, false);
    }

    // ---- Stage prefetched tile into the other buffer; one barrier per iter.
    if (more) {
      const int nb = buf ^ 1;
#pragma unroll
      for (int j = 0; j < 2; ++j) {
        int r = sr[j], c = sc4[j] * 4;
        Ks[nb][r][c + 0] = (_Float16)pk[j].x;
        Ks[nb][r][c + 1] = (_Float16)pk[j].y;
        Ks[nb][r][c + 2] = (_Float16)pk[j].z;
        Ks[nb][r][c + 3] = (_Float16)pk[j].w;
        Vst[nb][c + 0][r] = (_Float16)pv[j].x;
        Vst[nb][c + 1][r] = (_Float16)pv[j].y;
        Vst[nb][c + 2][r] = (_Float16)pv[j].z;
        Vst[nb][c + 3][r] = (_Float16)pv[j].w;
      }
      __syncthreads(); // writes to nb visible; reads of buf complete
    }
    buf ^= 1;
  }

  // ---- Normalize by row sum and store (O C-layout: q = v+8*hf, d = dt*16+ln16)
  // Output written exactly once -> nontemporal stores (don't rinse K/V from L2).
  float lr = 1.0f / lrun;
#pragma unroll
  for (int v = 0; v < 8; ++v) {
    float lv = __shfl(lr, (hf << 3) + v, 32);
    int qr = qbase + (hf << 3) + v;
    float* op = Og + ((size_t)b * SEQLEN + qr) * DK + ln16;
#pragma unroll
    for (int dt = 0; dt < 4; ++dt)
      __builtin_nontemporal_store(Oacc[dt][v] * lv, op + dt * 16);
  }
}

extern "C" void kernel_launch(void* const* d_in, const int* in_sizes, int n_in,
                              void* d_out, int out_size, void* d_ws, size_t ws_size,
                              hipStream_t stream) {
  (void)in_sizes; (void)n_in; (void)d_ws; (void)ws_size; (void)out_size;
  const float* q = (const float*)d_in[0];
  const float* k = (const float*)d_in[1];
  const float* v = (const float*)d_in[2];
  float* o = (float*)d_out;
  dim3 grid(SEQLEN / BQ, NBATCH);  // (16, 32)
  dim3 block(256);                 // 8 wave32s
  flash_attn_f16wmma<<<grid, block, 0, stream>>>(q, k, v, o);
}
